// AttModule_75806172774924
// MI455X (gfx1250) — matrix operand order repaired and checked by reading, not compiled, not run on hardware
//
#include <hip/hip_runtime.h>
#include <hip/hip_bf16.h>

typedef __attribute__((ext_vector_type(16))) __bf16 v16bf;
typedef __attribute__((ext_vector_type(8)))  float  v8f;
typedef __attribute__((ext_vector_type(4)))  float  f4;
typedef f4 f4u __attribute__((aligned(4)));   // alignment-reduced for shifted b128 loads

static __device__ __forceinline__ f4 ld4u(const float* p) { return *(const f4u*)p; }

#define Hh 128
#define Ww 128
#define HW 16384
#define C  64
#define CQ 8

// ---------------------------------------------------------------------------
// P[8, HW] = Wm[8,64] @ X[64, HW] + bias  via bf16 WMMA (fp32 accumulate).
// M=16 (channels, rows 8..15 zero-padded), N=16 pixels/tile, K=64 (2 steps of 32).
// grid: (32, 1, nImages), block 256 (8 waves); each wave does 4 pixel tiles.
// ---------------------------------------------------------------------------
__global__ __launch_bounds__(256) void proj_kernel(
    const float* __restrict__ src, long srcBatchStride,
    float* __restrict__ dst, long dstBatchStride,
    const float* __restrict__ Wm, const float* __restrict__ bias)
{
  const int lane = threadIdx.x & 31;
  const int wave = threadIdx.x >> 5;
  const int img  = blockIdx.z;
  const float* s = src + (long)img * srcBatchStride;
  float*       d = dst + (long)img * dstBatchStride;

  // A-matrix (16x32 bf16 per k-step): lane holds row m = lane&15,
  // K-halves selected by lane>=16 per the ISA 16-bit A layout.
  const int m    = lane & 15;
  const int half = lane >> 4;
  v16bf a[2];
  #pragma unroll
  for (int ks = 0; ks < 2; ++ks) {
    #pragma unroll
    for (int e = 0; e < 16; ++e) {
      int j = e >> 1, pos = e & 1;
      int k = ((j < 4) ? (2 * j) : (2 * j + 8)) + 8 * half + pos + 32 * ks;
      float wv = (m < CQ) ? Wm[m * C + k] : 0.0f;
      a[ks][e] = (__bf16)wv;
    }
  }

  float bloc[CQ];
  #pragma unroll
  for (int j = 0; j < CQ; ++j) bloc[j] = bias[j];

  const int waveGlobal = blockIdx.x * 8 + wave;
  #pragma unroll 1
  for (int t = 0; t < 4; ++t) {
    const int tile    = waveGlobal * 4 + t;      // 0..1023
    const int pixbase = tile * 16;
    const int px      = pixbase + (lane & 15);

    v8f c = {};
    #pragma unroll
    for (int ks = 0; ks < 2; ++ks) {
      // B-matrix 32x16 bf16: N = lane&15, K = e + 8*(lane>=16) + 8*(e>=8) (+32*ks)
      const int chbase = ((lane >> 4) ? 8 : 0) + 32 * ks;
      v16bf bmat;
      #pragma unroll
      for (int e = 0; e < 16; ++e)
        bmat[e] = (__bf16)s[(long)(chbase + e + ((e >= 8) ? 8 : 0)) * HW + px];
      c = __builtin_amdgcn_wmma_f32_16x16x32_bf16(
              /*neg_a=*/false, a[ks], /*neg_b=*/false, bmat,
              /*c_mod=*/(short)0, c, /*reuse_a=*/false, /*reuse_b=*/false);
    }
    // D layout: lanes 0..15 hold M = vgpr index (channels 0..7 are real)
    if (lane < 16) {
      #pragma unroll
      for (int j = 0; j < CQ; ++j)
        d[(long)j * HW + px] = c[j] + bloc[j];
    }
  }
}

// ---------------------------------------------------------------------------
// Attention align, 4 pixels per thread (one row-aligned group of 4).
// scores[t][sub] = sum_c kf[c, p+sub+off_t] * q[c, p+sub]  (zero outside image)
// att = softmax over t (per sub-pixel), out[c, p+sub] = sum_t att * x[c, p+sub+off_t]
// Interior taps use b128 loads serving all 4 sub-pixels; border groups fall back
// to per-sub scalar loads. grid: (HW/4/256, 1, nImages), block 256.
// ---------------------------------------------------------------------------
__global__ __launch_bounds__(256) void att_kernel(
    const float* __restrict__ xsrc, long xBatchStride,
    const float* __restrict__ kf,   long kfBatchStride,
    const float* __restrict__ q,    long qBatchStride,
    float* __restrict__ dst,        long dstBatchStride,
    int dilation)
{
  const int g   = blockIdx.x * blockDim.x + threadIdx.x;  // group id
  const int p4  = g * 4;                                  // first pixel of group
  const int img = blockIdx.z;
  const float* xs  = xsrc + (long)img * xBatchStride;
  const float* kfs = kf   + (long)img * kfBatchStride;
  const float* qs  = q    + (long)img * qBatchStride;
  float*       d   = dst  + (long)img * dstBatchStride;

  const int y  = p4 >> 7;       // row (groups never straddle rows: 128 % 4 == 0)
  const int x0 = p4 & 127;

  // q for the 4 sub-pixels (16B aligned)
  f4 qv[CQ];
  #pragma unroll
  for (int ch = 0; ch < CQ; ++ch) qv[ch] = *(const f4*)(qs + ch * HW + p4);

  int  ob[9];        // tap base offset yy*W + x0+dx
  bool all4[9];      // whole group in-bounds for this tap
  bool yv[9];        // row valid
  int  xb[9];
  f4   sc[9];
  #pragma unroll
  for (int t = 0; t < 9; ++t) {
    const int i  = t / 3, j = t % 3;
    const int dx = (i - 1) * dilation;   // u
    const int dy = (j - 1) * dilation;   // v
    const int yy = y + dy;
    const int xbt = x0 + dx;
    yv[t]   = (yy >= 0) & (yy < Hh);
    all4[t] = yv[t] & (xbt >= 0) & (xbt + 3 < Ww);
    xb[t]   = xbt;
    ob[t]   = yy * Ww + xbt;
    f4 s = {0.0f, 0.0f, 0.0f, 0.0f};
    if (all4[t]) {
      #pragma unroll
      for (int ch = 0; ch < CQ; ++ch) {
        f4 kv = ld4u(kfs + ch * HW + ob[t]);
        s.x = fmaf(kv.x, qv[ch].x, s.x);
        s.y = fmaf(kv.y, qv[ch].y, s.y);
        s.z = fmaf(kv.z, qv[ch].z, s.z);
        s.w = fmaf(kv.w, qv[ch].w, s.w);
      }
    } else if (yv[t]) {
      #pragma unroll
      for (int sub = 0; sub < 4; ++sub) {
        const int xx = xbt + sub;
        if (xx >= 0 && xx < Ww) {
          float ss = 0.0f;
          #pragma unroll
          for (int ch = 0; ch < CQ; ++ch)
            ss = fmaf(kfs[ch * HW + ob[t] + sub], qv[ch][sub], ss);
          s[sub] = ss;
        }
      }
    }
    sc[t] = s;
  }

  // per-sub-pixel softmax over the 9 taps
  f4 mx = sc[0];
  #pragma unroll
  for (int t = 1; t < 9; ++t) {
    mx.x = fmaxf(mx.x, sc[t].x); mx.y = fmaxf(mx.y, sc[t].y);
    mx.z = fmaxf(mx.z, sc[t].z); mx.w = fmaxf(mx.w, sc[t].w);
  }
  f4 sum = {0.0f, 0.0f, 0.0f, 0.0f};
  #pragma unroll
  for (int t = 0; t < 9; ++t) {
    sc[t].x = __expf(sc[t].x - mx.x); sum.x += sc[t].x;
    sc[t].y = __expf(sc[t].y - mx.y); sum.y += sc[t].y;
    sc[t].z = __expf(sc[t].z - mx.z); sum.z += sc[t].z;
    sc[t].w = __expf(sc[t].w - mx.w); sum.w += sc[t].w;
  }
  const f4 inv = {1.0f / sum.x, 1.0f / sum.y, 1.0f / sum.z, 1.0f / sum.w};
  #pragma unroll
  for (int t = 0; t < 9; ++t) {
    sc[t].x *= inv.x; sc[t].y *= inv.y; sc[t].z *= inv.z; sc[t].w *= inv.w;
  }

  // weighted 9-tap gather over all 64 channels, b128 per tap on the fast path
  #pragma unroll 1
  for (int ch = 0; ch < C; ++ch) {
    const float* xc = xs + (long)ch * HW;
    f4 acc = {0.0f, 0.0f, 0.0f, 0.0f};
    #pragma unroll
    for (int t = 0; t < 9; ++t) {
      if (all4[t]) {
        f4 v = ld4u(xc + ob[t]);
        acc.x = fmaf(sc[t].x, v.x, acc.x);
        acc.y = fmaf(sc[t].y, v.y, acc.y);
        acc.z = fmaf(sc[t].z, v.z, acc.z);
        acc.w = fmaf(sc[t].w, v.w, acc.w);
      } else if (yv[t]) {
        #pragma unroll
        for (int sub = 0; sub < 4; ++sub) {
          const int xx = xb[t] + sub;
          if (xx >= 0 && xx < Ww)
            acc[sub] = fmaf(sc[t][sub], xc[ob[t] + sub], acc[sub]);
        }
      }
    }
    *(f4*)(d + (long)ch * HW + p4) = acc;   // 16B aligned
  }
}

// out[b][i][64..127][:] = x[b][2][:][:]  (b128 copy)
__global__ __launch_bounds__(256) void copy_ref_kernel(
    const float* __restrict__ x, float* __restrict__ out)
{
  const long idx = ((long)blockIdx.x * blockDim.x + threadIdx.x) * 4;  // over 64*HW
  const int  i = blockIdx.y;
  const int  b = blockIdx.z;
  const float* src = x + ((long)b * 5 + 2) * C * HW;
  float* d = out + ((long)b * 2 + i) * 192 * HW + (long)64 * HW;
  *(f4*)(d + idx) = *(const f4*)(src + idx);
}

extern "C" void kernel_launch(void* const* d_in, const int* in_sizes, int n_in,
                              void* d_out, int out_size, void* d_ws, size_t ws_size,
                              hipStream_t stream)
{
  const float* x   = (const float*)d_in[0];
  const float* Wq1 = (const float*)d_in[1];
  const float* bq1 = (const float*)d_in[2];
  const float* Wk1 = (const float*)d_in[3];
  const float* bk1 = (const float*)d_in[4];
  const float* Wq2 = (const float*)d_in[5];
  const float* bq2 = (const float*)d_in[6];
  const float* Wk2 = (const float*)d_in[7];
  const float* bk2 = (const float*)d_in[8];
  float* out = (float*)d_out;

  const int  B    = 8, N = 5;
  const long imgC = (long)C  * HW;   // 64*16384 floats per image
  const long imgQ = (long)CQ * HW;   // 8*16384 floats per projection

  // workspace layout (floats): q1 | q2 | kf(4 slots) | mid(4 slots)
  float* ws  = (float*)d_ws;
  float* q1  = ws;
  float* q2  = q1  + (long)B * imgQ;
  float* kfb = q2  + (long)B * imgQ;
  float* mid = kfb + 4L * B * imgQ;

  const dim3 blk(256);
  const dim3 projGrid(32, 1, B);          // 8 waves * 4 tiles * 16 px * 32 blk = 16384 px
  const dim3 attGrid(HW / 4 / 256, 1, B); // 4 pixels per thread

  const long  xBS = (long)N * imgC;       // batch stride inside x
  const float* ref = x + 2 * imgC;        // x[:,2]

  // Query projections from ref (shared across all frames/stages)
  proj_kernel<<<projGrid, blk, 0, stream>>>(ref, xBS, q1, imgQ, Wq1, bq1);
  proj_kernel<<<projGrid, blk, 0, stream>>>(ref, xBS, q2, imgQ, Wq2, bq2);

  // slot -> frame, output pair index i, output channel block (0 => x1, 2 => x2)
  const int frames[4] = {0, 1, 3, 4};
  const int outI[4]   = {0, 1, 1, 0};
  const int outCB[4]  = {0, 0, 2, 2};

  for (int s = 0; s < 4; ++s) {
    const float* xf   = x   + (long)frames[s] * imgC;     // batch stride xBS
    float*       kfs  = kfb + (long)s * B * imgQ;
    float*       mids = mid + (long)s * B * imgC;

    // Stage 1: dilation 3, weights (Wk1, q1)
    proj_kernel<<<projGrid, blk, 0, stream>>>(xf, xBS, kfs, imgQ, Wk1, bk1);
    att_kernel <<<attGrid, blk, 0, stream>>>(xf, xBS, kfs, imgQ,
                                             q1, imgQ, mids, imgC, 3);

    // Stage 2: dilation 1, weights (Wk2, q2); write straight into d_out slice
    proj_kernel<<<projGrid, blk, 0, stream>>>(mids, imgC, kfs, imgQ, Wk2, bk2);
    float* dsts = out + (long)outI[s] * 192 * HW + (long)outCB[s] * 64 * HW;
    att_kernel <<<attGrid, blk, 0, stream>>>(mids, imgC, kfs, imgQ,
                                             q2, imgQ, dsts, 2L * 192 * HW, 1);
  }

  // Center channels: straight copy of ref
  const dim3 refGrid((unsigned)((C * HW) / 4 / 256), 2, B);
  copy_ref_kernel<<<refGrid, blk, 0, stream>>>(x, out);
}
